// GraphAttentionLayer_5102421148287
// MI455X (gfx1250) — matrix operand order repaired and checked by reading, not compiled, not run on hardware
//
#include <hip/hip_runtime.h>
#include <hip/hip_bf16.h>
#include <math.h>

// GAT layer for MI455X (gfx1250, wave32, WMMA), N=8192, F_IN=256, F_OUT=64.
//
// Roofline: dominant traffic = adj (8192^2 x 4B = 268 MB) -> 11.5us @ 23.3TB/s.
// Single pass over adj with a STATIC softmax normalizer:
//   m_i = lrelu(f1[i] + max_j f2[j]) >= lrelu(f1[i]+f2[j])  (lrelu monotone)
// so no online max/rescale is needed; exp(e-m) <= 1 always. att@Wh (8.6 GFLOP)
// runs on V_WMMA_F32_16X16X32_F16. The hot loop is LDS-free: Wh is stored
// TRANSPOSED in f16 so B fragments are two 16B global loads, and each
// lane reads its own adj row slices as b128s (one 128B line per row per tile,
// shared by all 4 waves of the block through WGP$).

#define GAT_N     8192
#define GAT_FIN   256
#define GAT_FOUT  64
#define GAT_ALPHA 0.2f
#define GAT_MASK  (-9000000000000000.0f)

typedef __attribute__((ext_vector_type(16))) _Float16 v16h;
typedef __attribute__((ext_vector_type(8)))  float    v8f;

// ---------------------------------------------------------------------------
// Kernel 1: WhhT(f16)[col][row] = (h @ W)^T. 16-row tile, 4 waves x 16 cols.
// ---------------------------------------------------------------------------
__global__ __launch_bounds__(128) void gat_gemm_wh(
    const float* __restrict__ h, const float* __restrict__ W,
    _Float16* __restrict__ WhhT)
{
  __shared__ __align__(16) float    s_a[16][32];        // 2 KB h tile
  __shared__ __align__(16) _Float16 s_w[32][GAT_FOUT];  // 4 KB W tile (f16)

  const int tid  = threadIdx.x;
  const int lane = tid & 31;
  const int wave = tid >> 5;          // 0..3 -> output cols 16*wave..+15
  const int r    = lane & 15;         // A row / B,C col within tile
  const int g    = lane >> 4;         // lane group
  const int row0 = blockIdx.x * 16;

  v8f acc = {};

  for (int k0 = 0; k0 < GAT_FIN; k0 += 32) {
    __syncthreads();
    {
      // h tile: 16 rows x 32 f32, one float4 per thread (coalesced)
      int rr = tid >> 3, cc = (tid & 7) * 4;
      *(float4*)&s_a[rr][cc] =
          *(const float4*)(h + (size_t)(row0 + rr) * GAT_FIN + k0 + cc);
      // W tile: 32 rows x 64 f32 -> f16
      int rw = tid >> 2, cw = (tid & 3) * 16;
      const float* src = W + (size_t)(k0 + rw) * GAT_FOUT + cw;
      #pragma unroll
      for (int q = 0; q < 16; ++q) s_w[rw][cw + q] = (_Float16)src[q];
    }
    __syncthreads();

    // A fragment (16x32 f16): lane group g holds K = 16g+0..15
    v16h A, B;
    #pragma unroll
    for (int t = 0; t < 16; ++t) {
      int k = 16 * g + t;
      A[t] = (_Float16)s_a[r][k];
    }
    // B fragment (32x16 f16): lane holds col, VGPR v holds K = 16g+2v, +1
    int col = (wave << 4) + r;
    #pragma unroll
    for (int v = 0; v < 8; ++v) {
      B[2 * v]     = s_w[16 * g + 2 * v][col];
      B[2 * v + 1] = s_w[16 * g + 2 * v + 1][col];
    }
    acc = __builtin_amdgcn_wmma_f32_16x16x32_f16(
        false, A, false, B, (short)0, acc, false, false);
  }

  // C/D layout: VGPR v -> M = v + 8g, N = lane&15. Store TRANSPOSED.
  #pragma unroll
  for (int v = 0; v < 8; ++v) {
    int M = v + 8 * g;
    WhhT[(size_t)((wave << 4) + r) * GAT_N + row0 + M] = (_Float16)acc[v];
  }
}

// ---------------------------------------------------------------------------
// Kernel 2: f1 = Wh@a1, f2 = Wh@a2 from WhhT. One wave per row.
// ---------------------------------------------------------------------------
__global__ __launch_bounds__(256) void gat_f12(
    const _Float16* __restrict__ WhhT, const float* __restrict__ a,
    float* __restrict__ f1, float* __restrict__ f2)
{
  const int lane = threadIdx.x & 31;
  const int wave = threadIdx.x >> 5;
  const int row  = blockIdx.x * 8 + wave;

  float w0 = (float)WhhT[(size_t)lane * GAT_N + row];
  float w1 = (float)WhhT[(size_t)(lane + 32) * GAT_N + row];
  float s1 = w0 * a[lane]      + w1 * a[lane + 32];
  float s2 = w0 * a[64 + lane] + w1 * a[96 + lane];
  #pragma unroll
  for (int off = 16; off > 0; off >>= 1) {
    s1 += __shfl_xor(s1, off, 32);
    s2 += __shfl_xor(s2, off, 32);
  }
  if (lane == 0) { f1[row] = s1; f2[row] = s2; }
}

// ---------------------------------------------------------------------------
// Kernel 3: gmax = max_j f2[j] (single block).
// ---------------------------------------------------------------------------
__global__ __launch_bounds__(256) void gat_gmax(
    const float* __restrict__ f2, float* __restrict__ gmax)
{
  __shared__ float s[8];
  const int t = threadIdx.x;
  float m = -INFINITY;
  for (int i = t; i < GAT_N; i += 256) m = fmaxf(m, f2[i]);
  #pragma unroll
  for (int off = 16; off > 0; off >>= 1) m = fmaxf(m, __shfl_xor(m, off, 32));
  if ((t & 31) == 0) s[t >> 5] = m;
  __syncthreads();
  if (t < 8) {
    float v = s[t];
    #pragma unroll
    for (int off = 4; off > 0; off >>= 1) v = fmaxf(v, __shfl_xor(v, off, 32));
    if (t == 0) *gmax = v;
  }
}

// ---------------------------------------------------------------------------
// Kernel 4: fused masked softmax (static normalizer) + att@Wh + ELU.
// LDS-free hot loop; 16-row tile per block, 4 waves x 16 output cols.
// ---------------------------------------------------------------------------
__global__ __launch_bounds__(128) void gat_attn(
    const int* __restrict__ adj, const _Float16* __restrict__ WhhT,
    const float* __restrict__ f1, const float* __restrict__ f2,
    const float* __restrict__ gmaxp, float* __restrict__ out)
{
  const int lane = threadIdx.x & 31;
  const int wave = threadIdx.x >> 5;
  const int r    = lane & 15;          // attention row (dup across lane groups)
  const int g    = lane >> 4;          // lane group -> K sub-runs
  const int row0 = blockIdx.x * 16;

  const float f1r = f1[row0 + r];
  const float xm  = f1r + *gmaxp;
  const float m   = fmaxf(xm, GAT_ALPHA * xm);   // static row normalizer

  const int col = (wave << 4) + r;
  const _Float16* bcol = WhhT + (size_t)col * GAT_N;       // Wh column
  const int*      arow = adj  + (size_t)(row0 + r) * GAT_N; // adj row

  float l = 0.0f;
  v8f   acc = {};

  for (int j0 = 0; j0 < GAT_N; j0 += 32) {
    const int ka = j0 + 8 * g;         // K run 0: 8g+0..7
    const int kb = j0 + 16 + 8 * g;    // K run 1: 16+8g+0..7

    int   av[16];
    float fv[16];
    *(int4*)&av[0]    = *(const int4*)(arow + ka);
    *(int4*)&av[4]    = *(const int4*)(arow + ka + 4);
    *(int4*)&av[8]    = *(const int4*)(arow + kb);
    *(int4*)&av[12]   = *(const int4*)(arow + kb + 4);
    *(float4*)&fv[0]  = *(const float4*)(f2 + ka);
    *(float4*)&fv[4]  = *(const float4*)(f2 + ka + 4);
    *(float4*)&fv[8]  = *(const float4*)(f2 + kb);
    *(float4*)&fv[12] = *(const float4*)(f2 + kb + 4);

    // prefetch the adj stream ~1KB ahead (global_prefetch_b8)
    if (j0 + 256 < GAT_N) __builtin_prefetch(arow + j0 + 256 + 8 * g, 0, 1);

    // B fragment: lane's column, K = {8g+0..7, 16+8g+0..7} -> two 16B loads
    union { v16h v; int4 q[2]; } B;
    B.q[0] = *(const int4*)(bcol + j0 + 8 * g);
    B.q[1] = *(const int4*)(bcol + j0 + 16 + 8 * g);

    // A fragment: p = exp(lrelu(f1+f2) - m) (masked), accumulate row sum
    v16h A;
    #pragma unroll
    for (int t = 0; t < 16; ++t) {
      float x  = f1r + fv[t];
      float lr = fmaxf(x, GAT_ALPHA * x);          // leaky_relu
      float e  = (av[t] > 0) ? lr : GAT_MASK;      // adjacency mask
      float p  = __expf(e - m);                    // <= 1, masked -> 0
      l += p;
      A[t] = (_Float16)p;
    }

    acc = __builtin_amdgcn_wmma_f32_16x16x32_f16(
        false, A, false, B.v, (short)0, acc, false, false);
  }

  // combine the lane pair's disjoint K-runs -> full row sum
  l += __shfl_xor(l, 16, 32);

  // finalize: divide by softmax denom, ELU, store fp32
  #pragma unroll
  for (int v = 0; v < 8; ++v) {
    int M = v + 8 * g;                 // C/D row for this VGPR
    float lM  = __shfl(l, M, 32);      // row sum lives on lane M
    float val = acc[v] / lM;
    val = (val > 0.0f) ? val : (__expf(val) - 1.0f);
    out[(size_t)(row0 + M) * GAT_FOUT + col] = val;
  }
}

// ---------------------------------------------------------------------------
extern "C" void kernel_launch(void* const* d_in, const int* in_sizes, int n_in,
                              void* d_out, int out_size, void* d_ws, size_t ws_size,
                              hipStream_t stream) {
  (void)in_sizes; (void)n_in; (void)out_size; (void)ws_size;
  const float* h   = (const float*)d_in[0];   // (8192, 256) f32
  const int*   adj = (const int*)  d_in[1];   // (8192, 8192) i32
  const float* W   = (const float*)d_in[2];   // (256, 64) f32
  const float* a   = (const float*)d_in[3];   // (128, 1) f32
  float* out = (float*)d_out;                 // (8192, 64) f32

  // workspace: WhhT f16 (1 MB) | f1 (32 KB) | f2 (32 KB) | gmax (4 B)
  _Float16* WhhT = (_Float16*)d_ws;
  float* f1   = (float*)((char*)d_ws + (size_t)GAT_N * GAT_FOUT * sizeof(_Float16));
  float* f2   = f1 + GAT_N;
  float* gmax = f2 + GAT_N;

  gat_gemm_wh<<<GAT_N / 16, 128, 0, stream>>>(h, W, WhhT);
  gat_f12<<<GAT_N / 8, 256, 0, stream>>>(WhhT, a, f1, f2);
  gat_gmax<<<1, 256, 0, stream>>>(f2, gmax);
  gat_attn<<<GAT_N / 16, 128, 0, stream>>>(adj, WhhT, f1, f2, gmax, out);
}
